// DecoderBlock_87935160418974
// MI455X (gfx1250) — matrix operand order repaired and checked by reading, not compiled, not run on hardware
//
#include <hip/hip_runtime.h>
#include <hip/hip_bf16.h>

typedef __attribute__((ext_vector_type(16))) _Float16     v16h;
typedef __attribute__((ext_vector_type(8)))  float        v8f;
typedef __attribute__((ext_vector_type(4)))  float        v4f;
typedef __attribute__((ext_vector_type(4)))  unsigned int v4u;

namespace {
constexpr int kB       = 8;
constexpr int kInCh    = 128;
constexpr int kOutCh   = 128;
constexpr int kT       = 16384;
constexpr int kN       = 256;     // N_STATE
constexpr int kT2      = kT / 2;  // 8192 (FACTOR=2)
constexpr int kL       = 64;      // time-chunk length for the scan
constexpr int kNChunk  = kT / kL; // 256
constexpr int kNChunk2 = kT2 / kL;// 128
constexpr float kEps   = 1e-5f;
}

// ---------------------------------------------------------------------------
// WMMA fragment loads from LDS (CDNA5 ISA 7.12.2, 16-bit operands), arranged
// so every lane reads contiguous 16B chunks -> ds_load_b128.
// A (16x32, MxK) from row-major [M][K] tile:
//   lane l: m=l&15, kb=(l&16)?8:0; elems 0..7 = K kb..kb+7, 8..15 = K 16+kb..
// B (32x16, KxN) from TRANSPOSED [N][K] tile:
//   lane l: n=l&15, kb=(l&16)?8:0; elems 0..7 = K kb..kb+7, 8..15 = K 16+kb..
// ---------------------------------------------------------------------------
__device__ __forceinline__ v16h frag_a16(const _Float16* __restrict__ p, int ld, int lane) {
  int m  = lane & 15;
  int kb = (lane & 16) ? 8 : 0;
  const _Float16* q = p + m * ld + kb;
  union { v4u u[2]; v16h h; } r;
  r.u[0] = *(const v4u*)(q);        // K = kb .. kb+7
  r.u[1] = *(const v4u*)(q + 16);   // K = 16+kb .. 16+kb+7
  return r.h;
}

__device__ __forceinline__ v16h frag_bT16(const _Float16* __restrict__ p, int ld, int lane) {
  int n  = lane & 15;
  int kb = (lane & 16) ? 8 : 0;
  const _Float16* q = p + n * ld + kb;
  union { v4u u[2]; v16h h; } r;
  r.u[0] = *(const v4u*)(q);        // K = kb .. kb+7
  r.u[1] = *(const v4u*)(q + 16);   // K = 16+kb .. 16+kb+7
  return r.h;
}

// ---------------------------------------------------------------------------
// Kernel 1: discretization + weight conversion (tiny, one block).
//   lam = -softplus(raw_lambda); A_d = exp(lam); scale = (A_d-1)/lam
//   Apow[k][n]  = A_d[n]^k, k=0..kL           (chunked-scan fixups)
//   BdhT[n][c]  = f16( B_c[c][n] * scale[n] ) (B^T for GEMM1 B operand)
//   ChT [o][n]  = f16( C_mat[n][o] )          (C^T for GEMM2 B operand)
//   WkT [o][f*128+c] = f16( W[o][2c+f] )      (K-permuted W for conv1x1)
// ---------------------------------------------------------------------------
__global__ __launch_bounds__(256) void k1_prep(
    const float* __restrict__ raw_lambda, const float* __restrict__ Bc,
    const float* __restrict__ Cm, const float* __restrict__ W,
    float* __restrict__ Apow, float* __restrict__ scale,
    _Float16* __restrict__ BdhT, _Float16* __restrict__ ChT,
    _Float16* __restrict__ WkT) {
  int tid = threadIdx.x;
  if (tid < kN) {
    float rl  = raw_lambda[tid];
    float sp  = (rl > 20.f) ? rl : log1pf(__expf(rl));  // softplus, stable
    float lam = -sp;                                     // < 0
    float Ad  = __expf(lam);                             // DT = 1
    float sc  = (fabsf(lam) > 1e-12f) ? (Ad - 1.f) / lam : 1.f;
    scale[tid] = sc;
    float p = 1.f;
    Apow[tid] = 1.f;
    for (int k = 1; k <= kL; ++k) { p *= Ad; Apow[k * kN + tid] = p; }
  }
  __syncthreads();
  for (int i = tid; i < kN * kInCh; i += 256) {
    int n = i >> 7, c = i & 127;
    BdhT[i] = (_Float16)(Bc[(size_t)c * kN + n] * scale[n]);
  }
  for (int i = tid; i < kOutCh * kN; i += 256) {
    int o = i >> 8, n = i & 255;
    ChT[i] = (_Float16)Cm[(size_t)n * kOutCh + o];
  }
  for (int i = tid; i < kOutCh * kN; i += 256) {
    int o = i >> 8, cc = i & 255;
    int f = cc >> 7, c = cc & 127;
    WkT[i] = (_Float16)W[(size_t)o * kN + 2 * c + f];
  }
}

// ---------------------------------------------------------------------------
// Kernel 2: v = x^T * B_d (WMMA, K=128) fused with per-chunk local scan.
// Block = (chunk, batch). 256 threads = 8 waves. Tile grid: 4 (t) x 16 (n).
// LDS stage A (operands) is aliased by stage B (v + f16 scan out) after sync.
// ---------------------------------------------------------------------------
__global__ __launch_bounds__(256) void k2_gemm1_scan(
    const float* __restrict__ x, const _Float16* __restrict__ BdhT,
    const float* __restrict__ Apow, _Float16* __restrict__ ssg,
    float* __restrict__ carry) {
  constexpr int LDA  = kInCh + 8;   // 136 (mult of 8 -> 16B-aligned frags)
  constexpr int LDBT = kInCh + 8;   // 136
  constexpr int SM_A = (kL * LDA + kN * LDBT) * 2;          // 87040
  constexpr int SM_B = kL * kN * 4 + kL * kN * 2;           // 98304
  constexpr int SM   = (SM_A > SM_B) ? SM_A : SM_B;
  __shared__ __align__(16) char smem[SM];
  _Float16* la   = (_Float16*)smem;                   // [kL][LDA]   x^T (f16)
  _Float16* lbT  = la + kL * LDA;                     // [kN][LDBT]  Bd^T (f16)
  float*    lv   = (float*)smem;                      // [kL][kN]    v (alias)
  _Float16* ls16 = (_Float16*)(smem + kL * kN * 4);   // [kL][kN]    scan out

  int tid = threadIdx.x;
  int b = blockIdx.y, ch = blockIdx.x;
  int t0 = ch * kL;
  const float* xb = x + (size_t)b * kInCh * kT;

  for (int i = tid; i < kL * kInCh; i += 256) {
    int tl = i & (kL - 1), c = i >> 6;                // coalesced over t
    la[tl * LDA + c] = (_Float16)xb[(size_t)c * kT + t0 + tl];
  }
  {
    const v4u* bsrc = (const v4u*)BdhT;               // [kN][kInCh] row-major
    for (int i4 = tid; i4 < (kN * kInCh) / 8; i4 += 256) {
      int n = i4 >> 4, c8 = (i4 & 15) * 8;
      *(v4u*)(lbT + n * LDBT + c8) = bsrc[i4];
    }
  }
  if (ch + 1 < kNChunk)  // gfx1250 global_prefetch_b8 for next chunk of x
    __builtin_prefetch(xb + (size_t)(tid & (kInCh - 1)) * kT + t0 + kL, 0, 1);
  __syncthreads();

  int lane = tid & 31, wave = tid >> 5;
  int mt  = wave >> 1;          // 0..3  (t tile)
  int nt0 = (wave & 1) * 8;     // 0 or 8 (n tile base, 8 tiles per wave)

  v16h af[4];
#pragma unroll
  for (int kk = 0; kk < 4; ++kk)
    af[kk] = frag_a16(la + (mt * 16) * LDA + kk * 32, LDA, lane);

  v8f acc[8];
  for (int j = 0; j < 8; ++j) {
    v8f c = {};
#pragma unroll
    for (int kk = 0; kk < 4; ++kk) {
      v16h bf = frag_bT16(lbT + (nt0 + j) * 16 * LDBT + kk * 32, LDBT, lane);
      c = __builtin_amdgcn_wmma_f32_16x16x32_f16(false, af[kk], false, bf,
                                                 (short)0, c, false, false);
    }
    acc[j] = c;
  }
  __syncthreads();  // all operand reads done -> safe to alias lv

  int cn = lane & 15;
  int rb = (lane & 16) ? 8 : 0;
  for (int j = 0; j < 8; ++j) {
    int nb = (nt0 + j) * 16 + cn;
#pragma unroll
    for (int r = 0; r < 8; ++r)
      lv[(mt * 16 + rb + r) * kN + nb] = acc[j][r];
  }
  __syncthreads();

  // Local scan: thread `tid` owns state column n = tid (zero initial state).
  {
    int n = tid;
    float Ad = Apow[kN + n];  // A^1
    float s = 0.f;
    for (int t = 0; t < kL; ++t) {
      s = fmaf(s, Ad, lv[t * kN + n]);
      ls16[t * kN + n] = (_Float16)s;
    }
    carry[((size_t)b * kNChunk + ch) * kN + n] = s;
  }
  __syncthreads();

  // Coalesced b128 writeout of the chunk's local states.
  {
    const v4u* src = (const v4u*)ls16;
    v4u* dst = (v4u*)(ssg + ((size_t)b * kT + t0) * kN);
    for (int i4 = tid; i4 < (kL * kN) / 8; i4 += 256) dst[i4] = src[i4];
  }
}

// ---------------------------------------------------------------------------
// Kernel 3: sequential carry scan across chunks (8 blocks x 256 threads).
// prevc[b][ch][n] = global state just before chunk ch.
// ---------------------------------------------------------------------------
__global__ __launch_bounds__(256) void k3_carry(
    const float* __restrict__ Apow, const float* __restrict__ carry,
    float* __restrict__ prevc) {
  int b = blockIdx.x, n = threadIdx.x;
  float AL = Apow[kL * kN + n];  // A^kL
  float p = 0.f;
  const float* cr = carry + (size_t)b * kNChunk * kN + n;
  float* pw = prevc + (size_t)b * kNChunk * kN + n;
  for (int ch = 0; ch < kNChunk; ++ch) {
    pw[(size_t)ch * kN] = p;
    p = fmaf(p, AL, cr[(size_t)ch * kN]);
  }
}

// ---------------------------------------------------------------------------
// Kernel 4: scan fixup + y = ss * C (WMMA, K=256) + LayerNorm + SiLU.
// Block = (chunk, batch). Tile grid 4 (t) x 8 (o). y buffer aliases operands.
// ---------------------------------------------------------------------------
__global__ __launch_bounds__(256) void k4_gemm2_ln(
    const _Float16* __restrict__ ssg, const float* __restrict__ Apow,
    const float* __restrict__ prevc, const _Float16* __restrict__ ChT,
    const float* __restrict__ gamma, const float* __restrict__ beta,
    _Float16* __restrict__ hg) {
  constexpr int LDA = kN + 8;       // 264
  constexpr int LDC = kN + 8;       // 264 (C^T rows: [o][n])
  constexpr int LDY = kOutCh + 4;   // 132
  constexpr int SM  = (kL * LDA + kOutCh * LDC) * 2;  // 101376 >= 64*132*4
  __shared__ __align__(16) char smem[SM];
  __shared__ float sgam[kOutCh];
  __shared__ float sbet[kOutCh];
  _Float16* sa  = (_Float16*)smem;      // [kL][LDA]      fixed-up ss (f16)
  _Float16* scm = sa + kL * LDA;        // [kOutCh][LDC]  C^T (f16)
  float*    sy  = (float*)smem;         // [kL][LDY]      y (aliases sa/scm)

  int tid = threadIdx.x;
  int b = blockIdx.y, ch = blockIdx.x;
  int t0 = ch * kL;
  const float* pc = prevc + ((size_t)b * kNChunk + ch) * kN;
  const _Float16* src = ssg + ((size_t)b * kT + t0) * kN;

  if (tid < kOutCh) { sgam[tid] = gamma[tid]; sbet[tid] = beta[tid]; }
  for (int i = tid; i < kL * kN; i += 256) {
    int tl = i >> 8, n = i & (kN - 1);
    float v = (float)src[i] + Apow[(tl + 1) * kN + n] * pc[n];  // s_t fixup
    sa[tl * LDA + n] = (_Float16)v;
  }
  {
    const v4u* csrc = (const v4u*)ChT;  // [kOutCh][kN] row-major
    for (int i4 = tid; i4 < (kOutCh * kN) / 8; i4 += 256) {
      int o = i4 >> 5, n8 = (i4 & 31) * 8;
      *(v4u*)(scm + o * LDC + n8) = csrc[i4];
    }
  }
  __syncthreads();

  int lane = tid & 31, wave = tid >> 5;
  int mt  = wave >> 1;
  int nt0 = (wave & 1) * 4;

  v16h af[8];
#pragma unroll
  for (int kk = 0; kk < 8; ++kk)
    af[kk] = frag_a16(sa + (mt * 16) * LDA + kk * 32, LDA, lane);

  v8f acc[4];
  for (int j = 0; j < 4; ++j) {
    v8f c = {};
#pragma unroll
    for (int kk = 0; kk < 8; ++kk) {
      v16h bf = frag_bT16(scm + (nt0 + j) * 16 * LDC + kk * 32, LDC, lane);
      c = __builtin_amdgcn_wmma_f32_16x16x32_f16(false, af[kk], false, bf,
                                                 (short)0, c, false, false);
    }
    acc[j] = c;
  }
  __syncthreads();

  int cn = lane & 15;
  int rb = (lane & 16) ? 8 : 0;
  for (int j = 0; j < 4; ++j) {
    int ob = (nt0 + j) * 16 + cn;
#pragma unroll
    for (int r = 0; r < 8; ++r)
      sy[(mt * 16 + rb + r) * LDY + ob] = acc[j][r];
  }
  __syncthreads();

  // LayerNorm over the 128 output channels + SiLU; one row per thread.
  if (tid < kL) {
    const float* yr = sy + tid * LDY;
    float mu = 0.f;
    for (int o = 0; o < kOutCh; ++o) mu += yr[o];
    mu *= (1.f / kOutCh);
    float var = 0.f;
    for (int o = 0; o < kOutCh; ++o) { float d = yr[o] - mu; var += d * d; }
    var *= (1.f / kOutCh);
    float rstd = rsqrtf(var + kEps);
    _Float16* dst = hg + ((size_t)b * kT + t0 + tid) * kOutCh;
    for (int o8 = 0; o8 < kOutCh; o8 += 8) {  // packed b128 stores
      union { v4u u; _Float16 h[8]; } pk;
#pragma unroll
      for (int j = 0; j < 8; ++j) {
        int o = o8 + j;
        float v = (yr[o] - mu) * rstd * sgam[o] + sbet[o];
        pk.h[j] = (_Float16)(v * (1.f / (1.f + __expf(-v))));  // SiLU
      }
      *(v4u*)(dst + o8) = pk.u;
    }
  }
}

// ---------------------------------------------------------------------------
// Kernel 5: resample (factor-2 time->channel) + conv1x1 (WMMA, K=256) + bias.
// K permuted as cc' = f*128+c, so A[t'][f*128+c] = h[2t'+f][c] (contiguous
// copies) and B^T = WkT[o][cc'] from kernel 1.
// ---------------------------------------------------------------------------
__global__ __launch_bounds__(256) void k5_conv(
    const _Float16* __restrict__ hg, const _Float16* __restrict__ WkT,
    const float* __restrict__ bias, float* __restrict__ out) {
  constexpr int LDA = kN + 8;  // 264
  constexpr int LDW = kN + 8;  // 264 (WkT rows: [o][cc'])
  __shared__ __align__(16) _Float16 sa[kL * LDA];
  __shared__ __align__(16) _Float16 sw[kOutCh * LDW];

  int tid = threadIdx.x;
  int b = blockIdx.y, ch = blockIdx.x;
  int t0p = ch * kL;
  const _Float16* hb = hg + ((size_t)b * kT + (size_t)2 * t0p) * kOutCh;

  {
    const v4u* hsrc = (const v4u*)hb;   // 128 rows x 128 halves, contiguous
    for (int i4 = tid; i4 < (2 * kL * kOutCh) / 8; i4 += 256) {
      int t = i4 >> 4, c8 = (i4 & 15) * 8;
      int tp = t >> 1, f = t & 1;
      *(v4u*)(sa + tp * LDA + f * kOutCh + c8) = hsrc[i4];
    }
    const v4u* wsrc = (const v4u*)WkT;  // [kOutCh][kN] row-major
    for (int i4 = tid; i4 < (kOutCh * kN) / 8; i4 += 256) {
      int o = i4 >> 5, cc8 = (i4 & 31) * 8;
      *(v4u*)(sw + o * LDW + cc8) = wsrc[i4];
    }
  }
  __syncthreads();

  int lane = tid & 31, wave = tid >> 5;
  int mt  = wave >> 1;
  int nt0 = (wave & 1) * 4;

  v16h af[8];
#pragma unroll
  for (int kk = 0; kk < 8; ++kk)
    af[kk] = frag_a16(sa + (mt * 16) * LDA + kk * 32, LDA, lane);

  int cn = lane & 15;
  int rb = (lane & 16) ? 8 : 0;
  for (int j = 0; j < 4; ++j) {
    v8f c = {};
#pragma unroll
    for (int kk = 0; kk < 8; ++kk) {
      v16h bf = frag_bT16(sw + (nt0 + j) * 16 * LDW + kk * 32, LDW, lane);
      c = __builtin_amdgcn_wmma_f32_16x16x32_f16(false, af[kk], false, bf,
                                                 (short)0, c, false, false);
    }
    int o = (nt0 + j) * 16 + cn;
    float bo = bias[o];
    float* op = out + ((size_t)b * kOutCh + o) * kT2 + t0p + mt * 16 + rb;
    v4f lo = { c[0] + bo, c[1] + bo, c[2] + bo, c[3] + bo };
    v4f hi = { c[4] + bo, c[5] + bo, c[6] + bo, c[7] + bo };
    *(v4f*)(op)     = lo;   // 8 consecutive t' per lane, 16B-aligned
    *(v4f*)(op + 4) = hi;
  }
}

// ---------------------------------------------------------------------------
extern "C" void kernel_launch(void* const* d_in, const int* in_sizes, int n_in,
                              void* d_out, int out_size, void* d_ws, size_t ws_size,
                              hipStream_t stream) {
  (void)in_sizes; (void)n_in; (void)out_size; (void)ws_size;
  const float* x    = (const float*)d_in[0];
  const float* rl   = (const float*)d_in[1];
  const float* Bc   = (const float*)d_in[2];
  const float* Cm   = (const float*)d_in[3];
  const float* gam  = (const float*)d_in[4];
  const float* bet  = (const float*)d_in[5];
  const float* W    = (const float*)d_in[6];
  const float* bias = (const float*)d_in[7];
  float* out = (float*)d_out;

  // Workspace carve-out (256B aligned regions). Total ~105 MB.
  char* ws = (char*)d_ws;
  auto carve = [&](size_t bytes) -> char* {
    char* p = ws;
    ws += (bytes + 255) & ~(size_t)255;
    return p;
  };
  float*    Apow  = (float*)carve((size_t)(kL + 1) * kN * 4);
  float*    scale = (float*)carve((size_t)kN * 4);
  _Float16* BdhT  = (_Float16*)carve((size_t)kN * kInCh * 2);
  _Float16* ChT   = (_Float16*)carve((size_t)kOutCh * kN * 2);
  _Float16* WkT   = (_Float16*)carve((size_t)kOutCh * kN * 2);
  float*    carry = (float*)carve((size_t)kB * kNChunk * kN * 4);
  float*    prevc = (float*)carve((size_t)kB * kNChunk * kN * 4);
  _Float16* ssg   = (_Float16*)carve((size_t)kB * kT * kN * 2);     // 67 MB
  _Float16* hg    = (_Float16*)carve((size_t)kB * kT * kOutCh * 2); // 34 MB

  k1_prep<<<1, 256, 0, stream>>>(rl, Bc, Cm, W, Apow, scale, BdhT, ChT, WkT);
  k2_gemm1_scan<<<dim3(kNChunk, kB), 256, 0, stream>>>(x, BdhT, Apow, ssg, carry);
  k3_carry<<<kB, kN, 0, stream>>>(Apow, carry, prevc);
  k4_gemm2_ln<<<dim3(kNChunk, kB), 256, 0, stream>>>(ssg, Apow, prevc, ChT, gam,
                                                     bet, hg);
  k5_conv<<<dim3(kNChunk2, kB), 256, 0, stream>>>(hg, WkT, bias, out);
}
